// NFLGraphTransformer_55722905699395
// MI455X (gfx1250) — matrix operand order repaired and checked by reading, not compiled, not run on hardware
//
#include <hip/hip_runtime.h>

// ---------------------------------------------------------------------------
// NFL graph transformer: 4x GATv2 + social pool, CDNA5 (gfx1250, wave32).
// Big GEMMs run on v_wmma_f32_16x16x32_f16 (f16 in, f32 accumulate).
// Social-pool weights staged into LDS via the Tensor Data Mover (TDM).
// Segment softmax / segment sums use L2 float atomics.
// ---------------------------------------------------------------------------

typedef _Float16 half_t;
typedef __attribute__((ext_vector_type(16))) _Float16 v16h;
typedef __attribute__((ext_vector_type(8)))  float    v8f;
typedef __attribute__((ext_vector_type(4)))  unsigned int u32x4;
typedef __attribute__((ext_vector_type(4)))  int i32x4;
typedef __attribute__((ext_vector_type(8)))  int i32x8;

#define NNODE 10000
#define NEDGE 320000
#define NHEAD 4
#define NCH   128
#define NLAY  4
#define DIN   7
#define EDIM  5
#define HC    512   // NHEAD * NCH

union FragU { v16h v; uint4 q[2]; };

#define WMMA_F16(acc, a, b) \
  (acc) = __builtin_amdgcn_wmma_f32_16x16x32_f16(false, (a), false, (b), (short)0, (acc), false, false)

static __device__ inline float wave_sum(float v) {
  #pragma unroll
  for (int m = 16; m > 0; m >>= 1) v += __shfl_xor(v, m, 32);
  return v;
}

// float atomic max via int punning (valid for IEEE ordering; init to -1e30f)
static __device__ inline void atomic_max_f(float* addr, float val) {
  if (val >= 0.0f) atomicMax((int*)addr, __float_as_int(val));
  else             atomicMin((unsigned int*)addr, __float_as_uint(val));
}

// ---------------------------------------------------------------------------
// Tensor Data Mover: 1-D tile of nelem f16 from global -> LDS.
// D# per CDNA5 ISA ch.8: group0 = {count|lds_addr|global_addr|type=2},
// group1 = {data_size=2B, tensor_dim0=tile_dim0=nelem, tensor_dim1=tile_dim1=1,
//           dim0_stride=nelem}; groups 2/3 zero (<=2D tensor).
static __device__ inline void tdm_load_to_lds(unsigned lds_off, const void* gsrc,
                                              unsigned nelem) {
  unsigned long long ga = (unsigned long long)(uintptr_t)gsrc;
  u32x4 g0 = {};
  g0[0] = 1u;                                            // count = 1 (valid)
  g0[1] = lds_off;                                       // lds_addr (bytes)
  g0[2] = (unsigned)ga;                                  // global_addr[31:0]
  g0[3] = (unsigned)((ga >> 32) & 0x01FFFFFFu) | (2u << 30); // gaddr[56:32], type=2
  i32x8 g1 = {};
  g1[0] = (int)(1u << 16);                               // data_size=1 (2B); wg_mask=0
  g1[1] = (int)((nelem & 0xFFFFu) << 16);                // tensor_dim0[15:0] @ [63:48]
  g1[2] = (int)((nelem >> 16) & 0xFFFFu) | (1 << 16);    // dim0[31:16]; tensor_dim1=1
  g1[3] = (int)((nelem & 0xFFFFu) << 16);                // tile_dim0 @ [127:112]
  g1[4] = 1;                                             // tile_dim1 = 1
  g1[5] = (int)nelem;                                    // tensor_dim0_stride[31:0]
  g1[6] = 0;
  g1[7] = 0;
  i32x4 z4 = {};
#if __clang_major__ >= 23
  i32x8 z8 = {};
  __builtin_amdgcn_tensor_load_to_lds(g0, g1, z4, z4, z8, 0);
#else
  __builtin_amdgcn_tensor_load_to_lds(g0, g1, z4, z4, 0);
#endif
}

// ---------------------------------------------------------------------------
__global__ void k_fill(float* __restrict__ p, float v, int n) {
  int i = blockIdx.x * 256 + threadIdx.x;
  if (i < n) p[i] = v;
}

// Bt[n*K + k] = W[k*Nout + n]  (f32 -> f16, transposed for WMMA B fragments)
__global__ void k_cvt_w_t(const float* __restrict__ W, half_t* __restrict__ Bt,
                          int K, int Nout) {
  int i = blockIdx.x * 256 + threadIdx.x;
  if (i < K * Nout) {
    int n = i / K, k = i - n * K;
    Bt[i] = (half_t)W[(size_t)k * Nout + n];
  }
}

// h = relu(x @ emb_W + emb_b); also emits f16 copy for WMMA A-matrix use
__global__ void k_embed(const float* __restrict__ x, const float* __restrict__ W,
                        const float* __restrict__ b, float* __restrict__ h,
                        half_t* __restrict__ h16) {
  int n = blockIdx.x;
  int c = threadIdx.x;
  float acc = b[c];
  #pragma unroll
  for (int k = 0; k < DIN; ++k) acc += x[n * DIN + k] * W[k * NCH + c];
  acc = acc > 0.f ? acc : 0.f;
  h[(size_t)n * NCH + c]   = acc;
  h16[(size_t)n * NCH + c] = (half_t)acc;
}

// ---------------------------------------------------------------------------
// Generic WMMA GEMM: C[M,Nout] = A16[M,K] @ B (B stored transposed [Nout,K]).
// One wave computes a 16x64 tile (4 accumulators); 8 waves per block.
__global__ __launch_bounds__(256) void k_gemm_f16(
    const half_t* __restrict__ A, const half_t* __restrict__ Bt,
    float* __restrict__ Cmat, int M, int K, int Nout) {
  const int lane = threadIdx.x & 31;
  const int wave = threadIdx.x >> 5;
  const int tilesN = Nout >> 6;
  const int totalTiles = (M >> 4) * tilesN;
  const int tile = blockIdx.x * 8 + wave;
  if (tile >= totalTiles) return;               // wave-uniform
  const int tm = tile / tilesN;
  const int tn = tile % tilesN;
  const int row = tm * 16 + (lane & 15);
  const int kbA = (lane >> 4) * 16;             // A: 0 or 16
  const int kbB = (lane >> 4) * 16;             // B: 0 or 16

  v8f acc0 = {}, acc1 = {}, acc2 = {}, acc3 = {};
  const half_t* arow = A + (size_t)row * K;
  const half_t* b0 = Bt + (size_t)(tn * 64 +  0 + (lane & 15)) * K;
  const half_t* b1 = Bt + (size_t)(tn * 64 + 16 + (lane & 15)) * K;
  const half_t* b2 = Bt + (size_t)(tn * 64 + 32 + (lane & 15)) * K;
  const half_t* b3 = Bt + (size_t)(tn * 64 + 48 + (lane & 15)) * K;

  for (int k0 = 0; k0 < K; k0 += 32) {
    __builtin_prefetch(arow + k0 + 32, 0, 1);
    FragU a;
    a.q[0] = *(const uint4*)(arow + k0 + kbA);
    a.q[1] = *(const uint4*)(arow + k0 + kbA + 8);
    FragU fb;
    fb.q[0] = *(const uint4*)(b0 + k0 + kbB);
    fb.q[1] = *(const uint4*)(b0 + k0 + kbB + 8);
    WMMA_F16(acc0, a.v, fb.v);
    fb.q[0] = *(const uint4*)(b1 + k0 + kbB);
    fb.q[1] = *(const uint4*)(b1 + k0 + kbB + 8);
    WMMA_F16(acc1, a.v, fb.v);
    fb.q[0] = *(const uint4*)(b2 + k0 + kbB);
    fb.q[1] = *(const uint4*)(b2 + k0 + kbB + 8);
    WMMA_F16(acc2, a.v, fb.v);
    fb.q[0] = *(const uint4*)(b3 + k0 + kbB);
    fb.q[1] = *(const uint4*)(b3 + k0 + kbB + 8);
    WMMA_F16(acc3, a.v, fb.v);
  }
  const int colBase = tn * 64 + (lane & 15);
  const int rowTop  = tm * 16 + ((lane < 16) ? 0 : 8);
  #pragma unroll
  for (int r = 0; r < 8; ++r) {
    float* crow = Cmat + (size_t)(rowTop + r) * Nout + colBase;
    crow[ 0] = acc0[r];
    crow[16] = acc1[r];
    crow[32] = acc2[r];
    crow[48] = acc3[r];
  }
}

// ---------------------------------------------------------------------------
// Per (edge, head) wave: logits = <leaky(xl[src]+xr[dst]+ea@We), att>;
// running segment max over dst via float atomic max.
__global__ __launch_bounds__(256) void k_edge_logits(
    const float* __restrict__ xl, const float* __restrict__ xr,
    const float* __restrict__ ea, const float* __restrict__ We,
    const float* __restrict__ att, const int* __restrict__ ei,
    float* __restrict__ logit, float* __restrict__ mx) {
  int lane = threadIdx.x & 31;
  int w = blockIdx.x * 8 + (threadIdx.x >> 5);
  if (w >= NEDGE * NHEAD) return;
  int e = w >> 2, head = w & 3;
  int s = ei[e], d = ei[NEDGE + e];
  float a0 = ea[e * EDIM + 0], a1 = ea[e * EDIM + 1], a2 = ea[e * EDIM + 2];
  float a3 = ea[e * EDIM + 3], a4 = ea[e * EDIM + 4];
  const float* xls = xl + (size_t)s * HC + head * NCH;
  const float* xrd = xr + (size_t)d * HC + head * NCH;
  const float* Wh  = We + head * NCH;
  const float* ath = att + head * NCH;
  float part = 0.f;
  #pragma unroll
  for (int j = 0; j < 4; ++j) {
    int c = lane + j * 32;
    float ev = a0 * Wh[c] + a1 * Wh[HC + c] + a2 * Wh[2 * HC + c]
             + a3 * Wh[3 * HC + c] + a4 * Wh[4 * HC + c];
    float m = xls[c] + xrd[c] + ev;
    m = (m > 0.f) ? m : 0.2f * m;
    part += m * ath[c];
  }
  float tot = wave_sum(part);
  if (lane == 0) {
    logit[w] = tot;
    atomic_max_f(&mx[(size_t)d * NHEAD + head], tot);
  }
}

// ex = exp(logit - mx[dst]); den[dst] += ex  (logit buffer overwritten with ex)
__global__ void k_edge_expden(const int* __restrict__ ei, float* __restrict__ logit,
                              const float* __restrict__ mx, float* __restrict__ den) {
  int idx = blockIdx.x * 256 + threadIdx.x;
  if (idx >= NEDGE * NHEAD) return;
  int e = idx >> 2, head = idx & 3;
  int d = ei[NEDGE + e];
  float ex = __expf(logit[idx] - mx[(size_t)d * NHEAD + head]);
  logit[idx] = ex;
  atomicAdd(&den[(size_t)d * NHEAD + head], ex);
}

// accum[dst] += alpha * xl[src]   (per (edge,head) wave, 4 ch/lane)
__global__ __launch_bounds__(256) void k_edge_scatter(
    const int* __restrict__ ei, const float* __restrict__ exv,
    const float* __restrict__ den, const float* __restrict__ xl,
    float* __restrict__ accum) {
  int lane = threadIdx.x & 31;
  int w = blockIdx.x * 8 + (threadIdx.x >> 5);
  if (w >= NEDGE * NHEAD) return;
  int e = w >> 2, head = w & 3;
  int s = ei[e], d = ei[NEDGE + e];
  float alpha = exv[w] / den[(size_t)d * NHEAD + head];
  const float* xls = xl + (size_t)s * HC + head * NCH;
  float* ac = accum + (size_t)d * HC + head * NCH;
  #pragma unroll
  for (int j = 0; j < 4; ++j) {
    int c = lane + j * 32;
    atomicAdd(&ac[c], alpha * xls[c]);
  }
}

// h = LN(relu(head_mean(accum)+b) + h); one wave per node; also refresh h16
__global__ __launch_bounds__(256) void k_node_update(
    const float* __restrict__ accum, const float* __restrict__ gb,
    const float* __restrict__ lng, const float* __restrict__ lnb,
    float* __restrict__ h, half_t* __restrict__ h16) {
  int lane = threadIdx.x & 31;
  int n = blockIdx.x * 8 + (threadIdx.x >> 5);
  if (n >= NNODE) return;
  const float* ac = accum + (size_t)n * HC;
  float t[4]; float ssum = 0.f;
  #pragma unroll
  for (int j = 0; j < 4; ++j) {
    int c = lane + j * 32;
    float g = 0.25f * (ac[c] + ac[NCH + c] + ac[2 * NCH + c] + ac[3 * NCH + c]) + gb[c];
    g = g > 0.f ? g : 0.f;
    float v = g + h[(size_t)n * NCH + c];
    t[j] = v; ssum += v;
  }
  float mean = wave_sum(ssum) * (1.f / NCH);
  float vs = 0.f;
  #pragma unroll
  for (int j = 0; j < 4; ++j) { float dd = t[j] - mean; vs += dd * dd; }
  float var = wave_sum(vs) * (1.f / NCH);
  float inv = rsqrtf(var + 1e-5f);
  #pragma unroll
  for (int j = 0; j < 4; ++j) {
    int c = lane + j * 32;
    float y = (t[j] - mean) * inv * lng[c] + lnb[c];
    h[(size_t)n * NCH + c]   = y;
    h16[(size_t)n * NCH + c] = (half_t)y;
  }
}

// cnt[src] += 1
__global__ void k_edge_count(const int* __restrict__ ei, float* __restrict__ cnt) {
  int e = blockIdx.x * 256 + threadIdx.x;
  if (e < NEDGE) atomicAdd(&cnt[ei[e]], 1.0f);
}

// ---------------------------------------------------------------------------
// Fused social pool: per 16-edge tile (one wave):
//  T = relu([h16[src]|h16[dst]] @ W1 + b1)        (WMMA K=256 -> LDS f16)
//  I = T @ W2 + b2 ; G = T @ gW + gb              (WMMA K=128 x2, B from LDS)
//  sp_s[src] += I * sigmoid(G)                    (atomic scatter)
// W2/gW (32 KB each) are DMAed once per workgroup into LDS by the TDM.
__global__ __launch_bounds__(256) void k_social(
    const half_t* __restrict__ h16, const int* __restrict__ ei,
    const half_t* __restrict__ W1t, const float* __restrict__ b1,
    const half_t* __restrict__ W2t, const float* __restrict__ b2,
    const half_t* __restrict__ gWt, const float* __restrict__ gb,
    float* __restrict__ sps) {
  __shared__ __align__(16) half_t ldsW2[NCH * NCH];
  __shared__ __align__(16) half_t ldsGW[NCH * NCH];
  __shared__ __align__(16) half_t ldsT[8][16 * NCH];
  __shared__ int ldsSrc[8][16];
  const int lane = threadIdx.x & 31;
  const int wave = threadIdx.x >> 5;

  // TDM: stage gate/value weights into LDS (issued once, by wave 0)
  if (wave == 0) {
    tdm_load_to_lds((unsigned)(uintptr_t)&ldsW2[0], W2t, NCH * NCH);
    tdm_load_to_lds((unsigned)(uintptr_t)&ldsGW[0], gWt, NCH * NCH);
    __builtin_amdgcn_s_wait_tensorcnt(0);
  }
  __syncthreads();

  const int tile = blockIdx.x * 8 + wave;
  if (tile >= NEDGE / 16) return;               // wave-uniform (exact division)
  const int erow = tile * 16 + (lane & 15);
  const int s_idx = ei[erow];
  const int d_idx = ei[NEDGE + erow];
  if (lane < 16) ldsSrc[wave][lane] = s_idx;
  const half_t* hs = h16 + (size_t)s_idx * NCH;
  const half_t* hd = h16 + (size_t)d_idx * NCH;
  const int kbA = (lane >> 4) * 16;
  const int kbB = (lane >> 4) * 16;
  const v8f vzero = {};

  // --- GEMM1: [16,256] @ W1 -> [16,128] ---
  v8f acc[8];
  #pragma unroll
  for (int t = 0; t < 8; ++t) acc[t] = vzero;
  for (int k0 = 0; k0 < 2 * NCH; k0 += 32) {
    FragU a;
    int p0 = k0 + kbA;
    int p1 = p0 + 8;
    const half_t* s0 = (p0 < NCH) ? (hs + p0) : (hd + p0 - NCH);
    const half_t* s1 = (p1 < NCH) ? (hs + p1) : (hd + p1 - NCH);
    a.q[0] = *(const uint4*)s0;
    a.q[1] = *(const uint4*)s1;
    #pragma unroll
    for (int t = 0; t < 8; ++t) {
      FragU b;
      const half_t* bp = W1t + (size_t)(t * 16 + (lane & 15)) * (2 * NCH) + k0 + kbB;
      b.q[0] = *(const uint4*)bp;
      b.q[1] = *(const uint4*)(bp + 8);
      WMMA_F16(acc[t], a.v, b.v);
    }
  }
  // relu + bias -> LDS f16 (becomes A-matrix for GEMM2/3)
  #pragma unroll
  for (int t = 0; t < 8; ++t) {
    int col = t * 16 + (lane & 15);
    float bb = b1[col];
    #pragma unroll
    for (int r = 0; r < 8; ++r) {
      int m = (lane < 16) ? r : (r + 8);
      float v = acc[t][r] + bb;
      ldsT[wave][m * NCH + col] = (half_t)(v > 0.f ? v : 0.f);
    }
  }

  // --- GEMM2 + gate GEMM: [16,128] @ {W2, gW}, B fragments from LDS ---
  v8f accI[8], accG[8];
  #pragma unroll
  for (int t = 0; t < 8; ++t) { accI[t] = vzero; accG[t] = vzero; }
  const half_t* trow = ldsT[wave] + (lane & 15) * NCH;
  for (int k0 = 0; k0 < NCH; k0 += 32) {
    FragU a;
    a.q[0] = *(const uint4*)(trow + k0 + kbA);
    a.q[1] = *(const uint4*)(trow + k0 + kbA + 8);
    #pragma unroll
    for (int t = 0; t < 8; ++t) {
      FragU b;
      const half_t* bp = &ldsW2[(t * 16 + (lane & 15)) * NCH + k0 + kbB];
      b.q[0] = *(const uint4*)bp;
      b.q[1] = *(const uint4*)(bp + 8);
      WMMA_F16(accI[t], a.v, b.v);
      const half_t* gp = &ldsGW[(t * 16 + (lane & 15)) * NCH + k0 + kbB];
      b.q[0] = *(const uint4*)gp;
      b.q[1] = *(const uint4*)(gp + 8);
      WMMA_F16(accG[t], a.v, b.v);
    }
  }
  // gated scatter-add to sp_s[src]
  #pragma unroll
  for (int t = 0; t < 8; ++t) {
    int col = t * 16 + (lane & 15);
    float bI = b2[col], bG = gb[col];
    #pragma unroll
    for (int r = 0; r < 8; ++r) {
      int m = (lane < 16) ? r : (r + 8);
      float I = accI[t][r] + bI;
      float G = accG[t][r] + bG;
      float gv = I * (1.0f / (1.0f + __expf(-G)));
      atomicAdd(&sps[(size_t)ldsSrc[wave][m] * NCH + col], gv);
    }
  }
}

// out = LN(h + sp_s / max(cnt,1))
__global__ __launch_bounds__(256) void k_final(
    const float* __restrict__ h, const float* __restrict__ sps,
    const float* __restrict__ cnt, const float* __restrict__ g,
    const float* __restrict__ b, float* __restrict__ out) {
  int lane = threadIdx.x & 31;
  int n = blockIdx.x * 8 + (threadIdx.x >> 5);
  if (n >= NNODE) return;
  float ic = 1.0f / fmaxf(cnt[n], 1.0f);
  float t[4]; float ssum = 0.f;
  #pragma unroll
  for (int j = 0; j < 4; ++j) {
    int c = lane + j * 32;
    float v = h[(size_t)n * NCH + c] + sps[(size_t)n * NCH + c] * ic;
    t[j] = v; ssum += v;
  }
  float mean = wave_sum(ssum) * (1.f / NCH);
  float vs = 0.f;
  #pragma unroll
  for (int j = 0; j < 4; ++j) { float dd = t[j] - mean; vs += dd * dd; }
  float var = wave_sum(vs) * (1.f / NCH);
  float inv = rsqrtf(var + 1e-5f);
  #pragma unroll
  for (int j = 0; j < 4; ++j) {
    int c = lane + j * 32;
    out[(size_t)n * NCH + c] = (t[j] - mean) * inv * g[c] + b[c];
  }
}

// ---------------------------------------------------------------------------
extern "C" void kernel_launch(void* const* d_in, const int* in_sizes, int n_in,
                              void* d_out, int out_size, void* d_ws, size_t ws_size,
                              hipStream_t stream) {
  (void)in_sizes; (void)n_in; (void)out_size; (void)ws_size;
  const float* x    = (const float*)d_in[0];
  const int*   ei   = (const int*)d_in[1];
  const float* ea   = (const float*)d_in[2];
  const float* embW = (const float*)d_in[3];
  const float* embB = (const float*)d_in[4];
  const float* gWl  = (const float*)d_in[5];
  const float* gWr  = (const float*)d_in[6];
  const float* gWe  = (const float*)d_in[7];
  const float* gatt = (const float*)d_in[8];
  const float* gatb = (const float*)d_in[9];
  const float* lng  = (const float*)d_in[10];
  const float* lnb  = (const float*)d_in[11];
  const float* spW1 = (const float*)d_in[12];
  const float* spb1 = (const float*)d_in[13];
  const float* spW2 = (const float*)d_in[14];
  const float* spb2 = (const float*)d_in[15];
  const float* spgW = (const float*)d_in[16];
  const float* spgb = (const float*)d_in[17];
  const float* fng  = (const float*)d_in[18];
  const float* fnb  = (const float*)d_in[19];
  float* out = (float*)d_out;

  char* base = (char*)d_ws;
  size_t off = 0;
  auto carve = [&](size_t bytes) -> void* {
    void* p = base + off;
    off += (bytes + 255) & ~(size_t)255;
    return p;
  };
  float*  h     = (float*) carve((size_t)NNODE * NCH * 4);
  half_t* h16   = (half_t*)carve((size_t)NNODE * NCH * 2);
  float*  xl    = (float*) carve((size_t)NNODE * HC * 4);
  float*  xr    = (float*) carve((size_t)NNODE * HC * 4);
  float*  logit = (float*) carve((size_t)NEDGE * NHEAD * 4);
  float*  mx    = (float*) carve((size_t)NNODE * NHEAD * 4);
  float*  den   = (float*) carve((size_t)NNODE * NHEAD * 4);
  float*  accum = (float*) carve((size_t)NNODE * HC * 4);
  float*  sps   = (float*) carve((size_t)NNODE * NCH * 4);
  float*  cnt   = (float*) carve((size_t)NNODE * 4);
  half_t* Wlt   = (half_t*)carve((size_t)HC * NCH * 2);
  half_t* Wrt   = (half_t*)carve((size_t)HC * NCH * 2);
  half_t* W1t   = (half_t*)carve((size_t)NCH * 2 * NCH * 2);
  half_t* W2t   = (half_t*)carve((size_t)NCH * NCH * 2);
  half_t* gWt   = (half_t*)carve((size_t)NCH * NCH * 2);

  k_embed<<<NNODE, NCH, 0, stream>>>(x, embW, embB, h, h16);

  const int nh = NNODE * NHEAD;
  const int eh = NEDGE * NHEAD;
  const int gemmTiles = (NNODE / 16) * (HC / 64);

  for (int i = 0; i < NLAY; ++i) {
    k_cvt_w_t<<<(NCH * HC + 255) / 256, 256, 0, stream>>>(gWl + (size_t)i * NCH * HC, Wlt, NCH, HC);
    k_cvt_w_t<<<(NCH * HC + 255) / 256, 256, 0, stream>>>(gWr + (size_t)i * NCH * HC, Wrt, NCH, HC);
    k_fill<<<(nh + 255) / 256, 256, 0, stream>>>(mx, -1e30f, nh);
    k_fill<<<(nh + 255) / 256, 256, 0, stream>>>(den, 0.f, nh);
    k_fill<<<(NNODE * HC + 255) / 256, 256, 0, stream>>>(accum, 0.f, NNODE * HC);
    k_gemm_f16<<<(gemmTiles + 7) / 8, 256, 0, stream>>>(h16, Wlt, xl, NNODE, NCH, HC);
    k_gemm_f16<<<(gemmTiles + 7) / 8, 256, 0, stream>>>(h16, Wrt, xr, NNODE, NCH, HC);
    k_edge_logits<<<(eh + 7) / 8, 256, 0, stream>>>(
        xl, xr, ea, gWe + (size_t)i * EDIM * HC, gatt + (size_t)i * NHEAD * NCH, ei, logit, mx);
    k_edge_expden<<<(eh + 255) / 256, 256, 0, stream>>>(ei, logit, mx, den);
    k_edge_scatter<<<(eh + 7) / 8, 256, 0, stream>>>(ei, logit, den, xl, accum);
    k_node_update<<<(NNODE + 7) / 8, 256, 0, stream>>>(
        accum, gatb + i * NCH, lng + i * NCH, lnb + i * NCH, h, h16);
  }

  k_cvt_w_t<<<(2 * NCH * NCH + 255) / 256, 256, 0, stream>>>(spW1, W1t, 2 * NCH, NCH);
  k_cvt_w_t<<<(NCH * NCH + 255) / 256, 256, 0, stream>>>(spW2, W2t, NCH, NCH);
  k_cvt_w_t<<<(NCH * NCH + 255) / 256, 256, 0, stream>>>(spgW, gWt, NCH, NCH);
  k_fill<<<(NNODE * NCH + 255) / 256, 256, 0, stream>>>(sps, 0.f, NNODE * NCH);
  k_fill<<<(NNODE + 255) / 256, 256, 0, stream>>>(cnt, 0.f, NNODE);
  k_edge_count<<<(NEDGE + 255) / 256, 256, 0, stream>>>(ei, cnt);
  k_social<<<((NEDGE / 16) + 7) / 8, 256, 0, stream>>>(
      h16, ei, W1t, spb1, W2t, spb2, gWt, spgb, sps);
  k_final<<<(NNODE + 7) / 8, 256, 0, stream>>>(h, sps, cnt, fng, fnb, out);
}
